// E61cTiedDecayCell_4887672783107
// MI455X (gfx1250) — matrix operand order repaired and checked by reading, not compiled, not run on hardware
//
#include <hip/hip_runtime.h>
#include <math.h>

// ---------------------------------------------------------------------------
// proj = einsum('tbd,ed->tbe', x, W) + b  (GEMM M=32768, N=1024, K=1024)
// then sequential gated-decay scan over t with h*silu(h) output.
//
// GEMM: bf16x3 split-precision WMMA (v_wmma_f32_16x16x32_bf16).
//  - 128x128 block tile, 8 waves arranged 4(m) x 2(n); each wave owns
//    2 m-tiles x 4 n-tiles  -> 24 ds_load_b128 per 24 WMMAs per k-step.
//  - register-level software pipeline: next k-tile's global loads issued
//    under current k-tile's WMMAs.
// Scan: proj lives in d_out's outputs region and is overwritten in place.
// ---------------------------------------------------------------------------

typedef __bf16 bf16x8  __attribute__((ext_vector_type(8)));
typedef __bf16 bf16x16 __attribute__((ext_vector_type(16)));
typedef float  v8f     __attribute__((ext_vector_type(8)));

#define TT   2048
#define BB   16
#define DD   1024
#define MM   (TT * BB)          // 32768 rows of x (t,b)

#define M_BLK  128
#define N_BLK  128
#define K_TILE 32
#define NKT    (DD / K_TILE)    // 32 k-tiles
#define LDSS   40               // ushorts per LDS row: 32 data + 8 pad

static __device__ __forceinline__ unsigned short f2bf_rne(float f) {
  unsigned u = __float_as_uint(f);
  unsigned r = 0x7FFFu + ((u >> 16) & 1u);
  return (unsigned short)((u + r) >> 16);
}
static __device__ __forceinline__ float bf2f(unsigned short h) {
  return __uint_as_float(((unsigned)h) << 16);
}
static __device__ __forceinline__ unsigned pack2(unsigned short a, unsigned short b) {
  return (unsigned)a | ((unsigned)b << 16);
}
static __device__ __forceinline__ uint2 cvt_hi4(const float4 v) {
  return make_uint2(pack2(f2bf_rne(v.x), f2bf_rne(v.y)),
                    pack2(f2bf_rne(v.z), f2bf_rne(v.w)));
}
static __device__ __forceinline__ uint2 cvt_lo4(const float4 v) {
  unsigned short hx = f2bf_rne(v.x), hy = f2bf_rne(v.y);
  unsigned short hz = f2bf_rne(v.z), hw = f2bf_rne(v.w);
  return make_uint2(pack2(f2bf_rne(v.x - bf2f(hx)), f2bf_rne(v.y - bf2f(hy))),
                    pack2(f2bf_rne(v.z - bf2f(hz)), f2bf_rne(v.w - bf2f(hw))));
}
static __device__ __forceinline__ bf16x16 cat8(bf16x8 a, bf16x8 b) {
  return __builtin_shufflevector(a, b, 0,1,2,3,4,5,6,7,8,9,10,11,12,13,14,15);
}

__global__ __launch_bounds__(256)
void gemm_proj_kernel(const float* __restrict__ x, const float* __restrict__ W,
                      const float* __restrict__ bias, float* __restrict__ proj) {
  __shared__ alignas(16) unsigned short sAh[M_BLK * LDSS];
  __shared__ alignas(16) unsigned short sAl[M_BLK * LDSS];
  __shared__ alignas(16) unsigned short sBh[N_BLK * LDSS];
  __shared__ alignas(16) unsigned short sBl[N_BLK * LDSS];

  const int tid   = threadIdx.x;
  const int wv    = tid >> 5;
  const int wm    = wv >> 1;           // 0..3 : 32-row strip (2 m-tiles)
  const int wn    = wv & 1;            // 0..1 : 64-col half  (4 n-tiles)
  const int lane  = tid & 31;
  const int lrow  = lane & 15;
  const int lhalf = lane >> 4;
  const int m_base = blockIdx.y * M_BLK;
  const int n_base = blockIdx.x * N_BLK;

  // staging addresses for this thread (4 float4 per matrix per k-tile)
  int srow[4], scol[4];
#pragma unroll
  for (int i = 0; i < 4; ++i) {
    const int f = tid + (i << 8);      // 0..1023 float4 slots
    srow[i] = f >> 3;                  // 8 float4 per 32-float row
    scol[i] = (f & 7) << 2;            // float offset within row
  }

  v8f acc[2][4];
#pragma unroll
  for (int mi = 0; mi < 2; ++mi)
#pragma unroll
    for (int ji = 0; ji < 4; ++ji)
      acc[mi][ji] = (v8f){0.f,0.f,0.f,0.f,0.f,0.f,0.f,0.f};

  // ---- prologue: fetch k-tile 0 into registers
  float4 rx[4], rw[4];
#pragma unroll
  for (int i = 0; i < 4; ++i) {
    rx[i] = *reinterpret_cast<const float4*>(x + (size_t)(m_base + srow[i]) * DD + scol[i]);
    rw[i] = *reinterpret_cast<const float4*>(W + (size_t)(n_base + srow[i]) * DD + scol[i]);
  }

  for (int kt = 0; kt < NKT; ++kt) {
    // ---- convert + store current k-tile into LDS (bf16 hi/lo limbs)
#pragma unroll
    for (int i = 0; i < 4; ++i) {
      const int lo = srow[i] * LDSS + scol[i];
      *reinterpret_cast<uint2*>(&sAh[lo]) = cvt_hi4(rx[i]);
      *reinterpret_cast<uint2*>(&sAl[lo]) = cvt_lo4(rx[i]);
      *reinterpret_cast<uint2*>(&sBh[lo]) = cvt_hi4(rw[i]);
      *reinterpret_cast<uint2*>(&sBl[lo]) = cvt_lo4(rw[i]);
    }
    __syncthreads();

    // ---- prefetch next k-tile (wrapped index keeps control flow uniform);
    //      these global loads fly underneath the WMMAs below.
    const int k1 = ((kt + 1) & (NKT - 1)) * K_TILE;
#pragma unroll
    for (int i = 0; i < 4; ++i) {
      rx[i] = *reinterpret_cast<const float4*>(x + (size_t)(m_base + srow[i]) * DD + k1 + scol[i]);
      rw[i] = *reinterpret_cast<const float4*>(W + (size_t)(n_base + srow[i]) * DD + k1 + scol[i]);
    }

    // ---- A fragments for the wave's two 16-row tiles
    //      (16-bit A layout: lane<16 K=0..7 & 16..23 ; lane>=16 K=8..15 & 24..31)
    bf16x16 a_hi[2], a_lo[2];
#pragma unroll
    for (int mi = 0; mi < 2; ++mi) {
      const int aoff = (wm * 32 + mi * 16 + lrow) * LDSS + lhalf * 8;
      a_hi[mi] = cat8(*reinterpret_cast<const bf16x8*>(&sAh[aoff]),
                      *reinterpret_cast<const bf16x8*>(&sAh[aoff + 16]));
      a_lo[mi] = cat8(*reinterpret_cast<const bf16x8*>(&sAl[aoff]),
                      *reinterpret_cast<const bf16x8*>(&sAl[aoff + 16]));
    }

#pragma unroll
    for (int ji = 0; ji < 4; ++ji) {
      // B layout: lane<16 holds K=0..7 & 16..23 of column N=lane, lane>=16 K=8..15 & 24..31
      const int boff = (wn * 64 + ji * 16 + lrow) * LDSS + lhalf * 8;
      const bf16x16 b_hi = cat8(*reinterpret_cast<const bf16x8*>(&sBh[boff]),
                                *reinterpret_cast<const bf16x8*>(&sBh[boff + 16]));
      const bf16x16 b_lo = cat8(*reinterpret_cast<const bf16x8*>(&sBl[boff]),
                                *reinterpret_cast<const bf16x8*>(&sBl[boff + 16]));
#pragma unroll
      for (int mi = 0; mi < 2; ++mi) {
        acc[mi][ji] = __builtin_amdgcn_wmma_f32_16x16x32_bf16(
            false, a_lo[mi], false, b_hi, (short)0, acc[mi][ji], false, false);
        acc[mi][ji] = __builtin_amdgcn_wmma_f32_16x16x32_bf16(
            false, a_hi[mi], false, b_lo, (short)0, acc[mi][ji], false, false);
        acc[mi][ji] = __builtin_amdgcn_wmma_f32_16x16x32_bf16(
            false, a_hi[mi], false, b_hi, (short)0, acc[mi][ji], false, false);
      }
    }
    __syncthreads();   // fragment reads done before next restage
  }

  // ---- epilogue: C/D layout -> M = r + 8*lhalf, N = lrow ; fold in bias
#pragma unroll
  for (int ji = 0; ji < 4; ++ji) {
    const int n_col = n_base + wn * 64 + ji * 16 + lrow;
    const float bv = bias[n_col];
#pragma unroll
    for (int mi = 0; mi < 2; ++mi) {
      const int m_row = m_base + wm * 32 + mi * 16 + lhalf * 8;
#pragma unroll
      for (int r = 0; r < 8; ++r)
        proj[(size_t)(m_row + r) * DD + n_col] = acc[mi][ji][r] + bv;
    }
  }
}

// Sequential scan over t; `out` initially holds proj and is overwritten in place.
__global__ __launch_bounds__(256)
void scan_kernel(float* __restrict__ out, float* __restrict__ h) {
  const int e  = blockIdx.x * blockDim.x + threadIdx.x;   // 0..B*D-1
  const int BD = BB * DD;
  if (e >= BD) return;
  float hv = 0.0f;
  h[e] = 0.0f;                                            // h0 = 0
  for (int t = 0; t < TT; ++t) {
    const size_t idx = (size_t)t * BD + e;
    const float p  = out[idx];
    const float a  = 1.0f / (1.0f + __expf(-p));          // sigmoid(proj)
    const float th = tanhf(p);
    hv = a * hv + (1.0f - a) * th;
    const float sg = 1.0f / (1.0f + __expf(-hv));         // sigmoid(h)
    out[idx] = hv * hv * sg;                              // h * silu(h)
    h[(size_t)(t + 1) * BD + e] = hv;
  }
}

extern "C" void kernel_launch(void* const* d_in, const int* in_sizes, int n_in,
                              void* d_out, int out_size, void* d_ws, size_t ws_size,
                              hipStream_t stream) {
  (void)in_sizes; (void)n_in; (void)out_size; (void)d_ws; (void)ws_size;
  const float* x = (const float*)d_in[0];   // [T,B,D]
  const float* W = (const float*)d_in[1];   // [D,D]
  const float* b = (const float*)d_in[2];   // [D]
  float* outp = (float*)d_out;                        // outputs [T,B,D]
  float* hp   = outp + (size_t)TT * BB * DD;          // h [T+1,B,D]

  dim3 grid(DD / N_BLK, MM / M_BLK);   // (8, 256), n fastest -> x-tile reuse in L2
  gemm_proj_kernel<<<grid, 256, 0, stream>>>(x, W, b, outp);
  scan_kernel<<<(BB * DD) / 256, 256, 0, stream>>>(outp, hp);
}
